// MultiHeadAttentionSimple_16114717295255
// MI455X (gfx1250) — matrix operand order repaired and checked by reading, not compiled, not run on hardware
//
#include <hip/hip_runtime.h>
#include <hip/hip_bf16.h>

// ---------- types ----------
typedef __bf16 v16bf __attribute__((ext_vector_type(16)));
typedef __bf16 v8bf  __attribute__((ext_vector_type(8)));
typedef float  v8f   __attribute__((ext_vector_type(8)));

#define B_  8
#define L_  2048
#define DM_ 512
#define H_  8
#define M_  (B_ * L_)          // 16384 rows (b,l)

__device__ __forceinline__ unsigned short f2bf(float x) {
    unsigned int u = __float_as_uint(x);
    u += 0x7FFFu + ((u >> 16) & 1u);   // round-to-nearest-even
    return (unsigned short)(u >> 16);
}

// ---------- generic f32 -> bf16 convert ----------
__global__ void f2bf_kernel(const float* __restrict__ src,
                            unsigned short* __restrict__ dst, int n) {
    int i = blockIdx.x * blockDim.x + threadIdx.x;
    if (i < n) dst[i] = f2bf(src[i]);
}

// W1kT[d][c] = bf16(W1[512+c][d])   (transposed bottom half of W1)
__global__ void w1kt_kernel(const float* __restrict__ W1,
                            unsigned short* __restrict__ W1kT) {
    int t = blockIdx.x * blockDim.x + threadIdx.x; // 262144
    int d = t >> 9, c = t & 511;
    W1kT[(size_t)d * 512 + c] = f2bf(W1[(size_t)(512 + c) * 512 + d]);
}

// ---------- WkEffT[h][d][e] = bf16( sum_c Wk[e, h*512+c] * W1k[c,d] ) ----------
// one wave per 16x16 tile; WMMA bf16, f32 accumulate
__global__ __launch_bounds__(32)
void wkeff_kernel(const unsigned short* __restrict__ WkBF,   // [512][4096]
                  const unsigned short* __restrict__ W1kT,   // [512 d][512 c]
                  unsigned short* __restrict__ WkEffT) {     // [8][512 d][512 e]
    const int e0 = blockIdx.x * 16, d0 = blockIdx.y * 16, h = blockIdx.z;
    const int lane = threadIdx.x;
    const int half = lane >> 4, idx = lane & 15;
    const __bf16* A  = (const __bf16*)WkBF;
    const __bf16* Bm = (const __bf16*)W1kT;
    const __bf16* arow = A + (size_t)(e0 + idx) * 4096 + h * 512;
    const __bf16* brow = Bm + (size_t)(d0 + idx) * 512 + half * 16;
    v8f acc = {};
    for (int k0 = 0; k0 < 512; k0 += 32) {
        v16bf a = *(const v16bf*)(arow + k0 + half * 16);
        v16bf b = *(const v16bf*)(brow + k0);
        acc = __builtin_amdgcn_wmma_f32_16x16x32_bf16(false, a, false, b,
                                                      (short)0, acc, false, false);
    }
    // C layout: VGPR r holds (half==0 ? M=r : M=r+8), N=idx. Store transposed (d-major).
    unsigned short* dst = WkEffT + ((size_t)h * 512 + (d0 + idx)) * 512 + e0 + half * 8;
    #pragma unroll
    for (int r = 0; r < 8; ++r) dst[r] = f2bf(acc[r]);
}

// bkEff[h][d] = sum_c bk[h*512+c] * W1[512+c][d]
__global__ void bkeff_kernel(const float* __restrict__ bk,
                             const float* __restrict__ W1,
                             float* __restrict__ bkEff) {
    int t = blockIdx.x * blockDim.x + threadIdx.x; // 4096
    int h = t >> 9, d = t & 511;
    float acc = 0.f;
    for (int c = 0; c < 512; ++c)
        acc += bk[h * 512 + c] * W1[(size_t)(512 + c) * 512 + d];
    bkEff[t] = acc;
}

// qh[b][j] = bq[j] + sum_e left[b][e] * Wq[e][j]     (j = h*512+c)
__global__ void qh_kernel(const float* __restrict__ left, const float* __restrict__ Wq,
                          const float* __restrict__ bq, float* __restrict__ qh) {
    int t = blockIdx.x * blockDim.x + threadIdx.x; // 32768
    int b = t >> 12, j = t & 4095;
    float acc = bq[j];
    for (int e = 0; e < 512; ++e)
        acc += left[b * 512 + e] * Wq[(size_t)e * 4096 + j];
    qh[t] = acc;
}

// hq[b][h][d] = sum_c qh[b][h*512+c] * W1[c][d]
__global__ void hq_kernel(const float* __restrict__ qh, const float* __restrict__ W1,
                          float* __restrict__ hq) {
    int t = blockIdx.x * blockDim.x + threadIdx.x; // 32768
    int b = t >> 12, i = t & 4095;
    int h = i >> 9, d = i & 511;
    float acc = 0.f;
    for (int c = 0; c < 512; ++c)
        acc += qh[b * 4096 + h * 512 + c] * W1[(size_t)c * 512 + d];
    hq[t] = acc;
}

// ---------- main fused GEMM + tanh + dot(W2) : scores[h][m] ----------
// hk[m, h, :] = rightBF[m,:] @ WkEffT[h].T  (K=512), epilogue fused in-register.
__global__ __launch_bounds__(256)
void scores_kernel(const unsigned short* __restrict__ rightBF,  // [16384][512]
                   const unsigned short* __restrict__ WkEffT,   // [8][512 d][512 e]
                   const float* __restrict__ bkEff,             // [8][512]
                   const float* __restrict__ hq,                // [8][8][512]
                   const float* __restrict__ W2,                // [512]
                   float* __restrict__ scoresWS) {              // [8][16384]
    __shared__ __align__(32) unsigned short Btile[16 * 512];    // 16 KB
    const int h    = blockIdx.y;
    const int wave = threadIdx.x >> 5;
    const int lane = threadIdx.x & 31;
    const int half = lane >> 4, idx = lane & 15;
    const int m0   = (blockIdx.x * 8 + wave) * 16;
    const int b    = m0 >> 11;

    // Preload all 16 A-fragments (K=512) for this wave's 16 rows.
    const __bf16* arow = (const __bf16*)rightBF + (size_t)(m0 + idx) * 512;
    v16bf afrag[16];
    #pragma unroll
    for (int kk = 0; kk < 16; ++kk) {
        int k0 = kk * 32;
        afrag[kk] = *(const v16bf*)(arow + k0 + half * 16);
    }

    float partial[8] = {0.f,0.f,0.f,0.f,0.f,0.f,0.f,0.f};
    const float* hqrow = hq + ((size_t)b * 8 + h) * 512;
    const float* bkrow = bkEff + (size_t)h * 512;

    for (int n0 = 0; n0 < 512; n0 += 16) {
        __syncthreads();
        // cooperatively stage B rows d0..d0+15 (16KB) into LDS
        const uint4* src = (const uint4*)(WkEffT + ((size_t)h * 512 + n0) * 512);
        uint4* dst = (uint4*)Btile;
        for (int i = threadIdx.x; i < 1024; i += 256) dst[i] = src[i];
        __syncthreads();

        const __bf16* brow = (const __bf16*)Btile + idx * 512 + half * 16;
        v8f acc = {};
        #pragma unroll
        for (int kk = 0; kk < 16; ++kk) {
            v16bf bfrag = *(const v16bf*)(brow + kk * 32);
            acc = __builtin_amdgcn_wmma_f32_16x16x32_bf16(false, afrag[kk], false, bfrag,
                                                          (short)0, acc, false, false);
        }
        const float hqv = hqrow[n0 + idx] + bkrow[n0 + idx]; // per-lane fixed N column
        const float w2v = W2[n0 + idx];
        #pragma unroll
        for (int r = 0; r < 8; ++r)
            partial[r] += w2v * tanhf(hqv + acc[r]);
    }

    // reduce over the 16 lanes of each half-wave (N columns)
    #pragma unroll
    for (int r = 0; r < 8; ++r) {
        float v = partial[r];
        v += __shfl_xor(v, 1, 32);
        v += __shfl_xor(v, 2, 32);
        v += __shfl_xor(v, 4, 32);
        v += __shfl_xor(v, 8, 32);
        partial[r] = v;
    }
    if (idx == 0) { // lane 0 -> rows m0..m0+7, lane 16 -> rows m0+8..m0+15
        float* srow = scoresWS + (size_t)h * M_ + m0 + half * 8;
        #pragma unroll
        for (int r = 0; r < 8; ++r) srow[r] = partial[r];
    }
}

// ---------- masked softmax over L, writes attn in (h,b,l) layout ----------
__global__ __launch_bounds__(256)
void softmax_kernel(const float* __restrict__ scoresWS, const int* __restrict__ mask,
                    float* __restrict__ attnOut) {
    const int hb = blockIdx.x;          // h*8 + b
    const int b  = hb & 7;
    const float* srow = scoresWS + (size_t)hb * L_;
    const int*   mrow = mask + (size_t)b * L_;
    __shared__ float red[256];
    const int t = threadIdx.x;
    float vals[8];
    float mx = -INFINITY;
    #pragma unroll
    for (int i = 0; i < 8; ++i) {
        int l = t + i * 256;
        float s = (mrow[l] == 0) ? -INFINITY : srow[l];
        vals[i] = s;
        mx = fmaxf(mx, s);
    }
    red[t] = mx; __syncthreads();
    for (int o = 128; o > 0; o >>= 1) { if (t < o) red[t] = fmaxf(red[t], red[t + o]); __syncthreads(); }
    mx = red[0]; __syncthreads();
    float sum = 0.f;
    #pragma unroll
    for (int i = 0; i < 8; ++i) { vals[i] = __expf(vals[i] - mx); sum += vals[i]; }
    red[t] = sum; __syncthreads();
    for (int o = 128; o > 0; o >>= 1) { if (t < o) red[t] += red[t + o]; __syncthreads(); }
    const float inv = 1.0f / red[0];
    #pragma unroll
    for (int i = 0; i < 8; ++i) attnOut[(size_t)hb * L_ + t + i * 256] = vals[i] * inv;
}

// rAtt[b][h][e] = sum_l attn[h][b][l] * right[b][l][e]
__global__ __launch_bounds__(512)
void wsum_kernel(const float* __restrict__ attn, const float* __restrict__ right,
                 float* __restrict__ rAtt) {
    const int bh = blockIdx.x;       // b*8 + h
    const int b = bh >> 3, h = bh & 7;
    const int e = threadIdx.x;
    const float* arow  = attn + ((size_t)h * 8 + b) * L_;
    const float* rbase = right + (size_t)b * L_ * 512 + e;
    float acc = 0.f;
    for (int l = 0; l < L_; ++l) acc += arow[l] * rbase[(size_t)l * 512];
    rAtt[(size_t)bh * 512 + e] = acc;
}

// attended[b][h][d] = bv[h*512+d] + sum_e rAtt[b][h][e] * Wv[e][h*512+d]
__global__ __launch_bounds__(512)
void attended_kernel(const float* __restrict__ rAtt, const float* __restrict__ Wv,
                     const float* __restrict__ bv, float* __restrict__ attended) {
    const int bh = blockIdx.x, h = bh & 7;
    const int d = threadIdx.x;
    const float* r = rAtt + (size_t)bh * 512;
    float acc = bv[h * 512 + d];
    for (int e = 0; e < 512; ++e) acc += r[e] * Wv[(size_t)e * 4096 + h * 512 + d];
    attended[(size_t)bh * 512 + d] = acc;
}

// out[b][j] = bfc[j] + sum_i attended[b][i] * Wfc[i][j]
__global__ __launch_bounds__(512)
void out_kernel(const float* __restrict__ attended, const float* __restrict__ Wfc,
                const float* __restrict__ bfc, float* __restrict__ out) {
    const int b = blockIdx.x, j = threadIdx.x;
    const float* a = attended + (size_t)b * 4096;
    float acc = bfc[j];
    for (int i = 0; i < 4096; ++i) acc += a[i] * Wfc[(size_t)i * 512 + j];
    out[(size_t)b * 512 + j] = acc;
}

extern "C" void kernel_launch(void* const* d_in, const int* in_sizes, int n_in,
                              void* d_out, int out_size, void* d_ws, size_t ws_size,
                              hipStream_t stream) {
    const float* left  = (const float*)d_in[0];
    const float* right = (const float*)d_in[1];
    const int*   mask  = (const int*)  d_in[2];
    const float* Wq    = (const float*)d_in[3];
    const float* bq    = (const float*)d_in[4];
    const float* Wk    = (const float*)d_in[5];
    const float* bk    = (const float*)d_in[6];
    const float* Wv    = (const float*)d_in[7];
    const float* bv    = (const float*)d_in[8];
    const float* W1    = (const float*)d_in[9];
    const float* W2    = (const float*)d_in[10];
    const float* Wfc   = (const float*)d_in[11];
    const float* bfc   = (const float*)d_in[12];

    float* out     = (float*)d_out;          // [8][512] (== (8,1,512))
    float* attnOut = out + B_ * DM_;         // [64][2048] (== (H*B, L, 1))

    char* w = (char*)d_ws;
    auto alloc = [&](size_t bytes) -> void* {
        void* p = (void*)w; w += (bytes + 255) & ~(size_t)255; return p;
    };
    unsigned short* rightBF = (unsigned short*)alloc((size_t)M_ * 512 * 2);   // 16 MB
    unsigned short* WkBF    = (unsigned short*)alloc((size_t)512 * 4096 * 2); // 4 MB
    unsigned short* W1kT    = (unsigned short*)alloc((size_t)512 * 512 * 2);
    unsigned short* WkEffT  = (unsigned short*)alloc((size_t)8 * 512 * 512 * 2);
    float* bkEff    = (float*)alloc(4096 * 4);
    float* qh       = (float*)alloc(32768 * 4);
    float* hq       = (float*)alloc(32768 * 4);
    float* scoresWS = (float*)alloc((size_t)H_ * M_ * 4);
    float* rAtt     = (float*)alloc(64 * 512 * 4);
    float* attended = (float*)alloc(64 * 512 * 4);

    const int nRight = M_ * 512;        // 8388608
    const int nWk    = 512 * 4096;      // 2097152
    f2bf_kernel<<<(nRight + 255) / 256, 256, 0, stream>>>(right, rightBF, nRight);
    f2bf_kernel<<<(nWk + 255) / 256, 256, 0, stream>>>(Wk, WkBF, nWk);
    w1kt_kernel<<<1024, 256, 0, stream>>>(W1, W1kT);
    wkeff_kernel<<<dim3(32, 32, 8), 32, 0, stream>>>(WkBF, W1kT, WkEffT);
    bkeff_kernel<<<16, 256, 0, stream>>>(bk, W1, bkEff);
    qh_kernel<<<128, 256, 0, stream>>>(left, Wq, bq, qh);
    hq_kernel<<<128, 256, 0, stream>>>(qh, W1, hq);
    scores_kernel<<<dim3(128, 8), 256, 0, stream>>>(rightBF, WkEffT, bkEff, hq, W2, scoresWS);
    softmax_kernel<<<64, 256, 0, stream>>>(scoresWS, mask, attnOut);
    wsum_kernel<<<64, 512, 0, stream>>>(attnOut, right, rAtt);
    attended_kernel<<<64, 512, 0, stream>>>(rAtt, Wv, bv, attended);
    out_kernel<<<8, 512, 0, stream>>>(attended, Wfc, bfc, out);
}
